// INT8Linear_39058432590542
// MI455X (gfx1250) — matrix operand order repaired and checked by reading, not compiled, not run on hardware
//
#include <hip/hip_runtime.h>
#include <hip/hip_bf16.h>
#include <stdint.h>

typedef __attribute__((ext_vector_type(16))) _Float16 v16h;
typedef __attribute__((ext_vector_type(8)))  _Float16 v8h;
typedef __attribute__((ext_vector_type(8)))  float    v8f;

#define IN_F    4096
#define OUT_F   16384
#define KSTEPS  (IN_F / 32)   // 128 WMMA k-steps
#define THREADS 256
#define WAVES   8             // wave32 -> 8 waves per block
#define XBYTES  131072        // x staged as f16 fragments: 128*2*32*16 B
#define LDSBYTES (XBYTES + 64)

// LDS: x[16,4096] converted bf16->f16 and swizzled into WMMA A-fragment order
//   [kstep(128)][chunk(2)][lane(32)] x 16 bytes
// chunk0 = A of lanes 0-15 (slot<16: K0..7 -> VGPR0-3, slot>=16: K8..15 -> VGPR4-7)
// chunk1 = A of lanes 16-31 (slot<16: K16..23 -> VGPR0-3, slot>=16: K24..31 -> VGPR4-7)
// followed by float rowsum[16]

extern "C" __global__ __launch_bounds__(THREADS)
void int8linear_wmma_kernel(const __hip_bfloat16* __restrict__ x,
                            const int8_t*        __restrict__ weight,
                            const float*         __restrict__ scale,
                            __hip_bfloat16*      __restrict__ out)
{
    extern __shared__ __align__(16) unsigned char smem[];
    uint4* ldsv    = (uint4*)smem;
    float* rowsumL = (float*)(smem + XBYTES);

    const int tid  = threadIdx.x;
    const int lane = tid & 31;   // wave32

    // ---- stage x: bf16 -> f16, swizzle to A-fragment layout, and row-sum ----
    {
        const uint4* xv   = (const uint4*)x;  // 8 bf16 per 16B chunk
        const int    sub16 = tid & 15;        // 16 threads per row
        const int    row   = tid >> 4;        // x row 0..15
        float rsum = 0.0f;
        #pragma unroll
        for (int j = 0; j < 32; ++j) {
            int g = (row << 9) + sub16 + (j << 4);   // chunk index (row-local)
            uint4 c = xv[g];
            uint32_t wi[4] = {c.x, c.y, c.z, c.w};
            uint32_t wo[4];
            #pragma unroll
            for (int q = 0; q < 4; ++q) {
                union { uint32_t u; float f; } lo, hi;
                lo.u = wi[q] << 16;                  // low  bf16 -> f32
                hi.u = wi[q] & 0xFFFF0000u;          // high bf16 -> f32
                rsum += lo.f + hi.f;
                union { _Float16 h[2]; uint32_t u; } p;
                p.h[0] = (_Float16)lo.f;
                p.h[1] = (_Float16)hi.f;
                wo[q] = p.u;
            }
            int k   = (g & 511) << 3;                // element offset in row
            int ks  = k >> 5;
            int sub = (k >> 3) & 3;
            int ln  = row + ((sub & 1) << 4);
            ldsv[(ks * 2 + (sub >> 1)) * 32 + ln] =
                make_uint4(wo[0], wo[1], wo[2], wo[3]);
        }
        // reduce the 16 per-row partials (they live in one half-wave)
        #pragma unroll
        for (int m = 8; m; m >>= 1) rsum += __shfl_xor(rsum, m, 32);
        if (sub16 == 0) rowsumL[row] = rsum;
    }
    __syncthreads();

    // ---- each wave computes one 16x16 output tile ----
    const int tile = blockIdx.x * WAVES + (tid >> 5);  // 0..1023
    const int col  = (tile << 4) + (lane & 15);        // output channel
    const int half = lane >> 4;                        // B K-half

    const int8_t* wrow = weight + (size_t)col * IN_F + half * 16;

    v8f acc = {};
    uint4 braw = *(const uint4*)wrow;                  // prefetch k-step 0
    const uint32_t MAGIC = 0x64006400u;                // f16 1024.0 bytes

    for (int ks = 0; ks < KSTEPS; ++ks) {
        uint4 cur = braw;
        int nk = (ks + 1 < KSTEPS) ? (ks + 1) : ks;
        braw = *(const uint4*)(wrow + nk * 32);        // double-buffered B load

        // A fragment: two conflict-free ds_load_b128
        v8h a0 = ((const v8h*)smem)[ks * 64 + half * 32 + (lane & 15)];
        v8h a1 = ((const v8h*)smem)[ks * 64 + half * 32 + 16 + (lane & 15)];
        v16h A;
        #pragma unroll
        for (int i = 0; i < 8; ++i) { A[i] = a0[i]; A[8 + i] = a1[i]; }

        // B fragment: byte v -> f16 (1152 + v), bias corrected in epilogue.
        // per dword: 1 xor + 2 v_perm  (no per-element cvt/add needed)
        uint32_t w0 = cur.x ^ 0x80808080u;
        uint32_t w1 = cur.y ^ 0x80808080u;
        uint32_t w2 = cur.z ^ 0x80808080u;
        uint32_t w3 = cur.w ^ 0x80808080u;
        union { v16h h; uint32_t u[8]; } B;
        B.u[0] = __builtin_amdgcn_perm(MAGIC, w0, 0x05010500u);
        B.u[1] = __builtin_amdgcn_perm(MAGIC, w0, 0x05030502u);
        B.u[2] = __builtin_amdgcn_perm(MAGIC, w1, 0x05010500u);
        B.u[3] = __builtin_amdgcn_perm(MAGIC, w1, 0x05030502u);
        B.u[4] = __builtin_amdgcn_perm(MAGIC, w2, 0x05010500u);
        B.u[5] = __builtin_amdgcn_perm(MAGIC, w2, 0x05030502u);
        B.u[6] = __builtin_amdgcn_perm(MAGIC, w3, 0x05010500u);
        B.u[7] = __builtin_amdgcn_perm(MAGIC, w3, 0x05030502u);

        acc = __builtin_amdgcn_wmma_f32_16x16x32_f16(
                  false, A, false, B.h, (short)0, acc, false, false);
    }

    // ---- epilogue: remove 1152 bias, apply channel scale, cast bf16 ----
    // C layout: VGPR r -> M = r + 8*(lane>=16), N = lane&15
    const float s     = scale[col];
    const int   mbase = half * 8;
    #pragma unroll
    for (int r = 0; r < 8; ++r) {
        float v = s * (acc[r] - 1152.0f * rowsumL[mbase + r]);
        out[(size_t)(mbase + r) * OUT_F + col] = __hip_bfloat16(v);
    }
}

extern "C" void kernel_launch(void* const* d_in, const int* in_sizes, int n_in,
                              void* d_out, int out_size, void* d_ws, size_t ws_size,
                              hipStream_t stream) {
    const __hip_bfloat16* x = (const __hip_bfloat16*)d_in[0];
    const int8_t*         w = (const int8_t*)d_in[1];
    const float*          s = (const float*)d_in[2];
    __hip_bfloat16*     out = (__hip_bfloat16*)d_out;

    dim3 grid(OUT_F / 16 / WAVES);   // 128 blocks, 8 waves each -> 1024 tiles
    dim3 block(THREADS);
    hipLaunchKernelGGL(int8linear_wmma_kernel, grid, block,
                       LDSBYTES, stream, x, w, s, out);
}
